// SelfAttentionModule_28295244546562
// MI455X (gfx1250) — matrix operand order repaired and checked by reading, not compiled, not run on hardware
//
#include <hip/hip_runtime.h>
#include <hip/hip_bf16.h>

// Problem constants
#define CC   256
#define CH   128
#define NN   4096
#define BB   8
#define LOG2E 1.44269504088896f
#define KSCL (0.0625f * LOG2E)          // fold 1/sqrt(C)=1/16 into exp2 argument

typedef __attribute__((ext_vector_type(4)))  float    v4f;
typedef __attribute__((ext_vector_type(8)))  float    v8f;
typedef __attribute__((ext_vector_type(16))) _Float16 v16h;

union FragH { v16h h; v4f f4[2]; };        // 32B: one 16x32 f16 A/B fragment per lane
union Pack8 { _Float16 e[8]; v4f f4; };    // 16B: 8 packed halves

static __device__ __forceinline__ v8f wmma16(v16h a, v16h b, v8f c) {
    return __builtin_amdgcn_wmma_f32_16x16x32_f16(false, a, false, b, (short)0, c,
                                                  false, false);
}
#define Z8 ((v8f){0.f, 0.f, 0.f, 0.f, 0.f, 0.f, 0.f, 0.f})

// ---------------------------------------------------------------------------
// Kernel 0: x[b,c,n] f32 -> xT[b,n,c] f16  (LDS 32x32 tile transpose)
// ---------------------------------------------------------------------------
__global__ __launch_bounds__(256) void xpose_kernel(const float* __restrict__ x,
                                                    _Float16* __restrict__ xT) {
    __shared__ float tile[32][33];
    const int b  = blockIdx.z;
    const int c0 = blockIdx.y * 32;
    const int n0 = blockIdx.x * 32;
    const int tx = threadIdx.x, ty = threadIdx.y;
#pragma unroll
    for (int s = 0; s < 4; ++s)
        tile[ty + s * 8][tx] = x[((size_t)(b * CC + c0 + ty + s * 8)) * NN + n0 + tx];
    __syncthreads();
#pragma unroll
    for (int s = 0; s < 4; ++s)
        xT[((size_t)(b * NN + n0 + ty + s * 8)) * CC + c0 + tx] =
            (_Float16)tile[tx][ty + s * 8];
}

// ---------------------------------------------------------------------------
// Kernel 1: projections via WMMA, cross-iteration double-buffered B frags.
// Double-buffer loads are UNCONDITIONAL (clamped index) so the outstanding
// load count is statically known on every CFG edge -> incremental waits.
// ---------------------------------------------------------------------------
__global__ __launch_bounds__(128, 1) void proj_kernel(
    const _Float16* __restrict__ xT,
    const float* __restrict__ Wq, const float* __restrict__ bq,
    const float* __restrict__ Wk, const float* __restrict__ bk,
    const float* __restrict__ Wv, const float* __restrict__ bv,
    _Float16* __restrict__ qT, _Float16* __restrict__ kT,
    _Float16* __restrict__ vF)
{
    const int ot  = blockIdx.x;            // 0..7 q, 8..15 k, 16..31 v
    const int b   = blockIdx.y;
    const int tid = threadIdx.x;
    const int wid = tid >> 5;
    const int lane = tid & 31, col = lane & 15, h = lane >> 4;

    const float* W; const float* bias; int o0; int mode; _Float16* dqk = nullptr;
    if (ot < 8)       { W = Wq; bias = bq; o0 = ot * 16;        mode = 0; dqk = qT; }
    else if (ot < 16) { W = Wk; bias = bk; o0 = (ot - 8) * 16;  mode = 0; dqk = kT; }
    else              { W = Wv; bias = bv; o0 = (ot - 16) * 16; mode = 1; }

    // A fragments: weight rows (f32 -> f16), resident (K = 256 -> 8 frags)
    FragH wa[8];
    const float* wrow = W + (size_t)(o0 + col) * CC;
#pragma unroll
    for (int dk = 0; dk < 8; ++dk) {
        v4f r0 = *(const v4f*)(wrow + dk * 32 + h * 8);
        v4f r1 = *(const v4f*)(wrow + dk * 32 + h * 8 + 4);
        v4f r2 = *(const v4f*)(wrow + dk * 32 + 16 + h * 8);
        v4f r3 = *(const v4f*)(wrow + dk * 32 + 16 + h * 8 + 4);
#pragma unroll
        for (int j = 0; j < 4; ++j) {
            wa[dk].h[j]      = (_Float16)r0[j];
            wa[dk].h[4 + j]  = (_Float16)r1[j];
            wa[dk].h[8 + j]  = (_Float16)r2[j];
            wa[dk].h[12 + j] = (_Float16)r3[j];
        }
    }
    float bv8[8];
#pragma unroll
    for (int r = 0; r < 8; ++r) bv8[r] = bias[o0 + h * 8 + r];

    auto loadX = [&](int nt, FragH* xb) {
        const _Float16* xrow = xT + ((size_t)(b * NN + nt * 16 + col)) * CC;
#pragma unroll
        for (int dk = 0; dk < 8; ++dk) {
            xb[dk].f4[0] = *(const v4f*)(xrow + dk * 32 + h * 8);
            xb[dk].f4[1] = *(const v4f*)(xrow + dk * 32 + 16 + h * 8);
        }
    };

    auto body = [&](int nt, FragH* cur, FragH* nxt) {
        // unconditional double-buffer load; last iteration harmlessly rewraps
        const int ntn = (nt + 4 < NN / 16) ? (nt + 4) : wid;
        loadX(ntn, nxt);
        v8f acc0 = Z8, acc1 = Z8;                    // two independent chains
#pragma unroll
        for (int dk = 0; dk < 8; dk += 2) {
            acc0 = wmma16(wa[dk].h,     cur[dk].h,     acc0);
            acc1 = wmma16(wa[dk + 1].h, cur[dk + 1].h, acc1);
        }
        const int n0 = nt * 16;
        if (mode == 0) {
            Pack8 pk;
#pragma unroll
            for (int r = 0; r < 8; ++r)
                pk.e[r] = (_Float16)(acc0[r] + acc1[r] + bv8[r]);
            *(v4f*)(dqk + ((size_t)(b * NN + n0 + col)) * CH + o0 + h * 8) = pk.f4;
        } else {
#pragma unroll
            for (int r = 0; r < 8; ++r)
                vF[((size_t)(b * CC + o0 + h * 8 + r)) * NN + n0 + col] =
                    (_Float16)(acc0[r] + acc1[r] + bv8[r]);
        }
    };

    FragH xbA[8], xbB[8];
    loadX(wid, xbA);                                // prologue
    for (int nt = wid; nt < NN / 16; nt += 8) {     // unrolled x2, swap buffers
        body(nt, xbA, xbB);
        body(nt + 4, xbB, xbA);
    }
}

// ---------------------------------------------------------------------------
// Kernel 2: flash attention. Block = (batch, 16-row i-tile), 256 threads,
// 8 waves = (4 j-streams) x (2 c-halves). Each wave owns 8 c-tiles
// (accO = 64 VGPRs), K fragments double-buffered across chunks (unconditional
// clamped loads), V fragments pipelined in pairs. Online softmax in
// raw-energy domain; cross-wave flash combine via LDS.
// ---------------------------------------------------------------------------
__global__ __launch_bounds__(256, 1) void attn_kernel(
    const _Float16* __restrict__ qT, const _Float16* __restrict__ kT,
    const _Float16* __restrict__ vF, const float* __restrict__ x,
    const float* __restrict__ gamma, float* __restrict__ out)
{
    const int b  = blockIdx.y;
    const int i0 = blockIdx.x * 16;
    const int tid = threadIdx.x;
    const int wid = tid >> 5;
    const int lane = tid & 31, col = lane & 15, h = lane >> 4;
    const int js    = wid & 3;     // j-stream (chunk % 4)
    const int chalf = wid >> 2;    // c half: 0 -> c 0..127, 1 -> c 128..255

    __shared__ __align__(16) _Float16 Pbuf[8][16 * 32];   // per-wave P staging
    __shared__ float Obuf[4][16][128];                    // combine staging / pass
    __shared__ float Mbuf[4][16];
    __shared__ float Lbuf[4][16];

    // Q A-fragments, resident (d = 128 -> 4 frags)
    FragH qa[4];
    {
        const _Float16* qb = qT + ((size_t)(b * NN + i0 + col)) * CH;
#pragma unroll
        for (int dk = 0; dk < 4; ++dk) {
            qa[dk].f4[0] = *(const v4f*)(qb + dk * 32 + h * 16);
            qa[dk].f4[1] = *(const v4f*)(qb + dk * 32 + h * 16 + 8);
        }
    }

    v8f accO[8];
#pragma unroll
    for (int t = 0; t < 8; ++t) accO[t] = Z8;
    float m8[8], l8[8];
#pragma unroll
    for (int r = 0; r < 8; ++r) { m8[r] = -1e30f; l8[r] = 0.f; }

    auto loadK = [&](int j0, FragH* kb) {
        const _Float16* kp = kT + ((size_t)(b * NN + j0 + col)) * CH + h * 16;
#pragma unroll
        for (int dk = 0; dk < 4; ++dk) {
            kb[2 * dk].f4[0]     = *(const v4f*)(kp + dk * 32);
            kb[2 * dk].f4[1]     = *(const v4f*)(kp + dk * 32 + 8);
            kb[2 * dk + 1].f4[0] = *(const v4f*)(kp + 16 * CH + dk * 32);
            kb[2 * dk + 1].f4[1] = *(const v4f*)(kp + 16 * CH + dk * 32 + 8);
        }
    };

    auto body = [&](int chunk, FragH* cur, FragH* nxt) {
        const int j0 = chunk * 32;
        // ---- S = Q K^T from pre-loaded fragments (two independent chains)
        v8f s0 = Z8, s1 = Z8;
#pragma unroll
        for (int dk = 0; dk < 4; ++dk) {
            s0 = wmma16(qa[dk].h, cur[2 * dk].h,     s0);
            s1 = wmma16(qa[dk].h, cur[2 * dk + 1].h, s1);
        }
        // ---- double-buffer: unconditional next-chunk K loads (clamped)
        const int cn = (chunk + 4 < NN / 32) ? (chunk + 4) : js;
        loadK(cn * 32, nxt);
        const int cp = (chunk + 8 < NN / 32) ? (chunk + 8) : js;
        __builtin_prefetch(kT + ((size_t)(b * NN + cp * 32 + lane)) * CH, 0, 3);

        // ---- first V pair issued early (hides under softmax + LDS)
        FragH vb0, vb1;
        {
            const _Float16* vp =
                vF + ((size_t)(b * CC + chalf * 128 + col)) * NN + j0 + h * 16;
            vb0.f4[0] = *(const v4f*)(vp);
            vb0.f4[1] = *(const v4f*)(vp + 8);
            vb1.f4[0] = *(const v4f*)(vp + 16 * NN);
            vb1.f4[1] = *(const v4f*)(vp + 16 * NN + 8);
        }

        // ---- online softmax (raw-energy domain)
        float rmax[8];
#pragma unroll
        for (int r = 0; r < 8; ++r) rmax[r] = fmaxf(s0[r], s1[r]);
#pragma unroll
        for (int mk = 1; mk <= 8; mk <<= 1)
#pragma unroll
            for (int r = 0; r < 8; ++r)
                rmax[r] = fmaxf(rmax[r], __shfl_xor(rmax[r], mk, 32));

        float p0[8], p1[8], rsum[8], alpha[8];
#pragma unroll
        for (int r = 0; r < 8; ++r) {
            float mn = fmaxf(m8[r], rmax[r]);
            alpha[r] = exp2f((m8[r] - mn) * KSCL);
            p0[r]    = exp2f((s0[r] - mn) * KSCL);
            p1[r]    = exp2f((s1[r] - mn) * KSCL);
            rsum[r]  = p0[r] + p1[r];
            m8[r]    = mn;
        }
#pragma unroll
        for (int mk = 1; mk <= 8; mk <<= 1)
#pragma unroll
            for (int r = 0; r < 8; ++r) rsum[r] += __shfl_xor(rsum[r], mk, 32);
#pragma unroll
        for (int r = 0; r < 8; ++r) l8[r] = l8[r] * alpha[r] + rsum[r];
#pragma unroll
        for (int t = 0; t < 8; ++t)
#pragma unroll
            for (int r = 0; r < 8; ++r) accO[t][r] *= alpha[r];

        // ---- P: C-layout -> A-fragment layout via per-wave LDS tile
        _Float16* pb = Pbuf[wid];
#pragma unroll
        for (int r = 0; r < 8; ++r) {
            int row = r + h * 8;
            pb[row * 32 + col]      = (_Float16)p0[r];
            pb[row * 32 + 16 + col] = (_Float16)p1[r];
        }
        FragH pa;                              // same-wave LDS RAW, in-order DS pipe
        pa.f4[0] = *(const v4f*)(pb + col * 32 + h * 16);
        pa.f4[1] = *(const v4f*)(pb + col * 32 + h * 16 + 8);

        // ---- O += P * V : pipelined pairs over this wave's 8 c-tiles
#pragma unroll
        for (int g = 0; g < 4; ++g) {
            FragH nb0, nb1;
            if (g < 3) {                       // compile-time guard (fully unrolled)
                const _Float16* vp =
                    vF + ((size_t)(b * CC + chalf * 128 + (2 * g + 2) * 16 + col)) * NN +
                    j0 + h * 16;
                nb0.f4[0] = *(const v4f*)(vp);
                nb0.f4[1] = *(const v4f*)(vp + 8);
                nb1.f4[0] = *(const v4f*)(vp + 16 * NN);
                nb1.f4[1] = *(const v4f*)(vp + 16 * NN + 8);
            }
            accO[2 * g]     = wmma16(pa.h, vb0.h, accO[2 * g]);
            accO[2 * g + 1] = wmma16(pa.h, vb1.h, accO[2 * g + 1]);
            if (g < 3) { vb0 = nb0; vb1 = nb1; }
        }
    };

    FragH kbA[8], kbB[8];
    loadK(js * 32, kbA);                          // prologue
    for (int chunk = js; chunk < NN / 32; chunk += 8) {   // unrolled x2, swap buffers
        body(chunk, kbA, kbB);
        body(chunk + 4, kbB, kbA);
    }

    // ---- cross-wave flash combine (stats identical across c-halves: wid<4 writes)
    if (col == 0 && wid < 4) {
#pragma unroll
        for (int r = 0; r < 8; ++r) {
            Mbuf[js][h * 8 + r] = m8[r];
            Lbuf[js][h * 8 + r] = l8[r];
        }
    }
    const float g = gamma[0];
    for (int pass = 0; pass < 2; ++pass) {
        if (chalf == pass) {
#pragma unroll
            for (int t = 0; t < 8; ++t)
#pragma unroll
                for (int r = 0; r < 8; ++r)
                    Obuf[js][r + h * 8][t * 16 + col] = accO[t][r];
        }
        __syncthreads();
        for (int e = tid; e < 16 * 128; e += 256) {
            int i = e & 15, cl = e >> 4;
            float mstar = Mbuf[0][i];
#pragma unroll
            for (int w = 1; w < 4; ++w) mstar = fmaxf(mstar, Mbuf[w][i]);
            float num = 0.f, den = 0.f;
#pragma unroll
            for (int w = 0; w < 4; ++w) {
                float f = exp2f((Mbuf[w][i] - mstar) * KSCL);
                num += Obuf[w][i][cl] * f;
                den += Lbuf[w][i] * f;
            }
            int c = pass * 128 + cl;
            size_t idx = ((size_t)(b * CC + c)) * NN + i0 + i;
            out[idx] = g * (num / den) + x[idx];
        }
        __syncthreads();
    }
}

// ---------------------------------------------------------------------------
extern "C" void kernel_launch(void* const* d_in, const int* in_sizes, int n_in,
                              void* d_out, int out_size, void* d_ws, size_t ws_size,
                              hipStream_t stream) {
    const float* x     = (const float*)d_in[0];
    const float* Wq    = (const float*)d_in[1];
    const float* bq    = (const float*)d_in[2];
    const float* Wk    = (const float*)d_in[3];
    const float* bk    = (const float*)d_in[4];
    const float* Wv    = (const float*)d_in[5];
    const float* bv    = (const float*)d_in[6];
    const float* gamma = (const float*)d_in[7];
    float* out = (float*)d_out;

    char* ws = (char*)d_ws;                         // needs 48 MB
    _Float16* xT = (_Float16*)(ws);                 // [B][N][256]  16 MB
    _Float16* qT = (_Float16*)(ws + (16u << 20));   // [B][N][128]   8 MB
    _Float16* kT = (_Float16*)(ws + (24u << 20));   // [B][N][128]   8 MB
    _Float16* vF = (_Float16*)(ws + (32u << 20));   // [B][256][N]  16 MB

    xpose_kernel<<<dim3(NN / 32, CC / 32, BB), dim3(32, 8), 0, stream>>>(x, xT);
    proj_kernel<<<dim3(32, BB), 128, 0, stream>>>(xT, Wq, bq, Wk, bk, Wv, bv, qT, kT, vF);
    attn_kernel<<<dim3(NN / 16, BB), 256, 0, stream>>>(qT, kT, vF, x, gamma, out);
}
